// AggregationWithoutEpinions_8392366096485
// MI455X (gfx1250) — matrix operand order repaired and checked by reading, not compiled, not run on hardware
//
#include <hip/hip_runtime.h>

// ---------------------------------------------------------------------------
// GAT-style edge attention aggregation for MI455X (gfx1250, wave32, WMMA).
// D=128, N_SRC=N_DST=50000, E=640000.
// fp32 WMMA (16x16x4) for node/out GEMMs (software-pipelined K loop),
// bf16 WMMA (16x16x32) for the big per-edge attention MLP.
// ---------------------------------------------------------------------------

#define DD        128
#define N_SRCC    50000
#define N_DSTT    50000
#define N_EDGE    640000
#define NEG_SLOPE 0.01f

typedef __attribute__((ext_vector_type(2)))  float  v2f;
typedef __attribute__((ext_vector_type(8)))  float  v8f;
typedef __attribute__((ext_vector_type(16))) __bf16 v16bf;
typedef __attribute__((ext_vector_type(8)))  __bf16 bf8x;

__device__ __forceinline__ float lrelu_f(float x) { return x > 0.f ? x : NEG_SLOPE * x; }

// ---------------------------------------------------------------------------
// Pre-layout a fp32 128x128 weight matrix into f32 WMMA B-fragment order:
// wl[(ks*32 + lane)*16 + j*2 + p] = W[(ks*4 + (lane>>4)*2 + p)*128
//                                     + j*16 + (lane&15)]
// ---------------------------------------------------------------------------
__global__ void layout_w_f32_k(const float* __restrict__ W,
                               float* __restrict__ wl) {
    const int t = blockIdx.x * blockDim.x + threadIdx.x;
    if (t >= 32 * 32 * 16) return;
    const int p    = t & 1;
    const int j    = (t >> 1) & 7;
    const int lane = (t >> 4) & 31;
    const int ks   = t >> 9;
    const int k = ks * 4 + (lane >> 4) * 2 + p;
    const int n = j * 16 + (lane & 15);
    wl[t] = W[(size_t)k * DD + n];
}

// ---------------------------------------------------------------------------
// C[M x 128] = A[M x 128] @ W[128 x 128] + bias, fp32 WMMA 16x16x4.
// W supplied pre-laid-out (wl). One wave computes a 16-row strip.
// K loop software-pipelined: next K-step's loads issue under current WMMAs.
// ---------------------------------------------------------------------------
__global__ void gemm128_f32_wmma(const float* __restrict__ A,
                                 const float* __restrict__ wl,
                                 const float* __restrict__ bias,
                                 float* __restrict__ C,
                                 __bf16* __restrict__ Cbf,
                                 int M, int relu_out) {
    const int lane = threadIdx.x & 31;
    const int wid  = threadIdx.x >> 5;
    const int tile = blockIdx.x * 4 + wid;
    const int ntiles = (M + 15) >> 4;
    if (tile >= ntiles) return;                  // wave-uniform exit

    const int m    = lane & 15;                  // col within 16 (B/C/D), row for A
    const int half = lane >> 4;                  // 0: lanes 0-15, 1: lanes 16-31

    int arow_i = tile * 16 + m;
    if (arow_i >= M) arow_i = 0;                 // clamp; garbage rows never stored
    const float* __restrict__ arow = A + (size_t)arow_i * DD;

    v8f acc[8];
#pragma unroll
    for (int j = 0; j < 8; ++j) {
        const float bj = bias[j * 16 + m];
#pragma unroll
        for (int r = 0; r < 8; ++r) acc[j][r] = bj;
    }

    // Prologue: load K-step 0.
    v2f a_c;
    a_c.x = arow[half * 2];
    a_c.y = arow[half * 2 + 1];
    const float* __restrict__ wpc = wl + ((size_t)lane << 4);
    float4 w0 = *(const float4*)(wpc);
    float4 w1 = *(const float4*)(wpc + 4);
    float4 w2 = *(const float4*)(wpc + 8);
    float4 w3 = *(const float4*)(wpc + 12);

#pragma unroll
    for (int ks = 0; ks < 32; ++ks) {
        // Prefetch K-step ks+1 into a second live register set.
        v2f a_n = a_c;
        float4 n0 = w0, n1 = w1, n2 = w2, n3 = w3;
        if (ks < 31) {
            const int ka = (ks + 1) * 4 + half * 2;
            a_n.x = arow[ka];
            a_n.y = arow[ka + 1];
            const float* __restrict__ wp =
                wl + ((size_t)((ks + 1) * 32 + lane) << 4);
            n0 = *(const float4*)(wp);
            n1 = *(const float4*)(wp + 4);
            n2 = *(const float4*)(wp + 8);
            n3 = *(const float4*)(wp + 12);
        }

        v2f b;
        b.x = w0.x; b.y = w0.y;
        acc[0] = __builtin_amdgcn_wmma_f32_16x16x4_f32(false, a_c, false, b, (short)0, acc[0], false, false);
        b.x = w0.z; b.y = w0.w;
        acc[1] = __builtin_amdgcn_wmma_f32_16x16x4_f32(false, a_c, false, b, (short)0, acc[1], false, false);
        b.x = w1.x; b.y = w1.y;
        acc[2] = __builtin_amdgcn_wmma_f32_16x16x4_f32(false, a_c, false, b, (short)0, acc[2], false, false);
        b.x = w1.z; b.y = w1.w;
        acc[3] = __builtin_amdgcn_wmma_f32_16x16x4_f32(false, a_c, false, b, (short)0, acc[3], false, false);
        b.x = w2.x; b.y = w2.y;
        acc[4] = __builtin_amdgcn_wmma_f32_16x16x4_f32(false, a_c, false, b, (short)0, acc[4], false, false);
        b.x = w2.z; b.y = w2.w;
        acc[5] = __builtin_amdgcn_wmma_f32_16x16x4_f32(false, a_c, false, b, (short)0, acc[5], false, false);
        b.x = w3.x; b.y = w3.y;
        acc[6] = __builtin_amdgcn_wmma_f32_16x16x4_f32(false, a_c, false, b, (short)0, acc[6], false, false);
        b.x = w3.z; b.y = w3.w;
        acc[7] = __builtin_amdgcn_wmma_f32_16x16x4_f32(false, a_c, false, b, (short)0, acc[7], false, false);

        // Rotate (pure renaming after full unroll).
        a_c = a_n;
        w0 = n0; w1 = n1; w2 = n2; w3 = n3;
    }

#pragma unroll
    for (int j = 0; j < 8; ++j) {
        const int col = j * 16 + m;
#pragma unroll
        for (int r = 0; r < 8; ++r) {
            const int row_c = tile * 16 + half * 8 + r;
            if (row_c < M) {
                const float raw = acc[j][r];
                C[(size_t)row_c * DD + col] = relu_out ? lrelu_f(raw) : raw;
                if (Cbf) Cbf[(size_t)row_c * DD + col] = (__bf16)raw;
            }
        }
    }
}

// ---------------------------------------------------------------------------
// Init: n_f = 0, m = -inf, denom = 0
// ---------------------------------------------------------------------------
__global__ void init_state_k(float* __restrict__ mx, float* __restrict__ dn,
                             float* __restrict__ nf) {
    const int t = blockIdx.x * blockDim.x + threadIdx.x;
    if (t < N_DSTT * DD) nf[t] = 0.f;
    if (t < N_DSTT) { mx[t] = -__builtin_inff(); dn[t] = 0.f; }
}

// ---------------------------------------------------------------------------
// Pre-layout W_a1 (fp32 128x128) into bf16 WMMA B-fragment order.
// ---------------------------------------------------------------------------
__global__ void layout_wa1_k(const float* __restrict__ W_a1,
                             __bf16* __restrict__ wlay) {
    const int t = blockIdx.x * blockDim.x + threadIdx.x;
    if (t >= 4 * 8 * 32 * 16) return;
    const int p    = t & 15;
    const int lane = (t >> 4) & 31;
    const int j    = (t >> 9) & 7;
    const int kt   = t >> 12;
    const int k = kt * 32 + ((lane >> 4) * 8) + (p & 7) + ((p >> 3) * 16);
    const int n = j * 16 + (lane & 15);
    wlay[t] = (__bf16)W_a1[(size_t)k * DD + n];
}

// ---------------------------------------------------------------------------
// Edge scores: one wave per 16 edges.
//   e   = hs[src] + hd[dst]          (bf16 gather, A-fragment layout)
//   h1  = lrelu(e) @ W_a1 + b_a1     (4x8 = 32 v_wmma_f32_16x16x32_bf16)
//   s   = lrelu(h1) @ W_a2 + b_a2    (per-lane FMA + shfl_xor reduce)
// ---------------------------------------------------------------------------
__global__ void edge_scores_k(const __bf16* __restrict__ hsb,
                              const __bf16* __restrict__ hdb,
                              const int* __restrict__ src,
                              const int* __restrict__ dst,
                              const __bf16* __restrict__ wlay,
                              const float* __restrict__ b_a1,
                              const float* __restrict__ W_a2,
                              const float* __restrict__ b_a2,
                              float* __restrict__ s_out) {
    const int lane = threadIdx.x & 31;
    const int wid  = threadIdx.x >> 5;
    const int tile = blockIdx.x * 4 + wid;
    if (tile >= N_EDGE / 16) return;             // wave-uniform

    const int m    = lane & 15;
    const int half = lane >> 4;
    const int hi8  = half * 8;

    const int edge = tile * 16 + m;              // E divisible by 16: always valid
    const int si = src[edge];
    const int di = dst[edge];
    const __bf16* __restrict__ hr = hsb + (size_t)si * DD;
    const __bf16* __restrict__ dr = hdb + (size_t)di * DD;

    v16bf afr[4];
#pragma unroll
    for (int kt = 0; kt < 4; ++kt) {
        const int b0 = kt * 32 + hi8;
        const bf8x h0 = *(const bf8x*)(hr + b0);
        const bf8x d0 = *(const bf8x*)(dr + b0);
        const bf8x h1 = *(const bf8x*)(hr + b0 + 16);
        const bf8x d1 = *(const bf8x*)(dr + b0 + 16);
#pragma unroll
        for (int p = 0; p < 8; ++p) {
            afr[kt][p]     = (__bf16)lrelu_f((float)h0[p] + (float)d0[p]);
            afr[kt][8 + p] = (__bf16)lrelu_f((float)h1[p] + (float)d1[p]);
        }
    }

    v8f acc[8];
#pragma unroll
    for (int j = 0; j < 8; ++j) {
        const float bj = b_a1[j * 16 + m];
#pragma unroll
        for (int r = 0; r < 8; ++r) acc[j][r] = bj;
    }

#pragma unroll
    for (int kt = 0; kt < 4; ++kt) {
#pragma unroll
        for (int j = 0; j < 8; ++j) {
            const v16bf bfr =
                *(const v16bf*)(wlay + (size_t)(((kt * 8 + j) * 32 + lane) * 16));
            acc[j] = __builtin_amdgcn_wmma_f32_16x16x32_bf16(
                false, afr[kt], false, bfr, (short)0, acc[j], false, false);
        }
    }

    // Second MLP layer: 128 -> 1 dot product with W_a2.
    float p_[8];
#pragma unroll
    for (int r = 0; r < 8; ++r) p_[r] = 0.f;
#pragma unroll
    for (int j = 0; j < 8; ++j) {
        const float w2 = W_a2[j * 16 + m];
#pragma unroll
        for (int r = 0; r < 8; ++r) p_[r] += lrelu_f(acc[j][r]) * w2;
    }
#pragma unroll
    for (int r = 0; r < 8; ++r) {
        float v = p_[r];
        v += __shfl_xor(v, 1, 32);
        v += __shfl_xor(v, 2, 32);
        v += __shfl_xor(v, 4, 32);
        v += __shfl_xor(v, 8, 32);               // reduce within each 16-lane half
        p_[r] = v;
    }
    if (m == 0) {                                // lanes 0 and 16 write 8 rows each
        const float b2 = b_a2[0];
#pragma unroll
        for (int r = 0; r < 8; ++r)
            s_out[tile * 16 + half * 8 + r] = p_[r] + b2;
    }
}

// ---------------------------------------------------------------------------
// Segment max over dst (sign-split integer atomic trick on fp32 bits).
// ---------------------------------------------------------------------------
__global__ void seg_max_k(const float* __restrict__ s, const int* __restrict__ dst,
                          float* __restrict__ mx) {
    const int t = blockIdx.x * blockDim.x + threadIdx.x;
    if (t >= N_EDGE) return;
    const float v = s[t];
    const int d = dst[t];
    if (v >= 0.f) atomicMax((int*)(mx + d), __float_as_int(v));
    else          atomicMin((unsigned int*)(mx + d), __float_as_uint(v));
}

// ---------------------------------------------------------------------------
// ex = exp(s - m[dst]); denom[dst] += ex
// ---------------------------------------------------------------------------
__global__ void seg_exp_k(const float* __restrict__ s, const int* __restrict__ dst,
                          const float* __restrict__ mx, float* __restrict__ ex,
                          float* __restrict__ dn) {
    const int t = blockIdx.x * blockDim.x + threadIdx.x;
    if (t >= N_EDGE) return;
    const int d = dst[t];
    const float v = expf(s[t] - mx[d]);
    ex[t] = v;
    atomicAdd(dn + d, v);
}

// ---------------------------------------------------------------------------
// n_f[dst] += alpha * (hs[src] + hd[dst]); one wave per edge, float4 lanes.
// ---------------------------------------------------------------------------
__global__ void edge_aggregate_k(const float* __restrict__ hs,
                                 const float* __restrict__ hd,
                                 const int* __restrict__ src,
                                 const int* __restrict__ dst,
                                 const float* __restrict__ ex,
                                 const float* __restrict__ dn,
                                 float* __restrict__ nf) {
    const int tid  = blockIdx.x * blockDim.x + threadIdx.x;
    const int edge = tid >> 5;
    if (edge >= N_EDGE) return;
    const int lane = tid & 31;
    const int c    = lane * 4;
    const int si = src[edge];
    const int di = dst[edge];
    const float alpha = ex[edge] / dn[di];
    const float4 a = *(const float4*)(hs + (size_t)si * DD + c);
    const float4 b = *(const float4*)(hd + (size_t)di * DD + c);
    float* o = nf + (size_t)di * DD + c;
    atomicAdd(o + 0, alpha * (a.x + b.x));
    atomicAdd(o + 1, alpha * (a.y + b.y));
    atomicAdd(o + 2, alpha * (a.z + b.z));
    atomicAdd(o + 3, alpha * (a.w + b.w));
}

// ---------------------------------------------------------------------------
extern "C" void kernel_launch(void* const* d_in, const int* in_sizes, int n_in,
                              void* d_out, int out_size, void* d_ws, size_t ws_size,
                              hipStream_t stream) {
    const float* feat_src = (const float*)d_in[0];
    const float* feat_dst = (const float*)d_in[1];
    const int*   src_idx  = (const int*)d_in[2];
    const int*   dst_idx  = (const int*)d_in[3];
    const float* W_src    = (const float*)d_in[4];
    const float* b_src    = (const float*)d_in[5];
    const float* W_dst    = (const float*)d_in[6];
    const float* b_dst    = (const float*)d_in[7];
    const float* W_a1     = (const float*)d_in[8];
    const float* b_a1     = (const float*)d_in[9];
    const float* W_a2     = (const float*)d_in[10];
    const float* b_a2     = (const float*)d_in[11];
    const float* W_out    = (const float*)d_in[12];
    const float* b_out    = (const float*)d_in[13];

    // Workspace layout (~103.2 MB); offsets keep 16/32B alignment.
    float* ws     = (float*)d_ws;
    float* hs     = ws;                                  // N_SRC*D f32
    float* hd     = hs + (size_t)N_SRCC * DD;            // N_DST*D f32
    float* nf     = hd + (size_t)N_DSTT * DD;            // N_DST*D f32
    float* sc     = nf + (size_t)N_DSTT * DD;            // E f32
    float* ex     = sc + N_EDGE;                         // E f32
    float* mx     = ex + N_EDGE;                         // N_DST f32
    float* dn     = mx + N_DSTT;                         // N_DST f32
    float* wl_src = dn + N_DSTT;                         // 16384 f32 (64KB)
    float* wl_dst = wl_src + 16384;                      // 16384 f32
    float* wl_out = wl_dst + 16384;                      // 16384 f32
    __bf16* hsb  = (__bf16*)(wl_out + 16384);            // N_SRC*D bf16
    __bf16* hdb  = hsb + (size_t)N_SRCC * DD;            // N_DST*D bf16
    __bf16* wlay = hdb + (size_t)N_DSTT * DD;            // 16384 bf16

    // 1. init state + weight fragment layouts
    init_state_k<<<(N_DSTT * DD + 255) / 256, 256, 0, stream>>>(mx, dn, nf);
    layout_w_f32_k<<<64, 256, 0, stream>>>(W_src, wl_src);
    layout_w_f32_k<<<64, 256, 0, stream>>>(W_dst, wl_dst);
    layout_w_f32_k<<<64, 256, 0, stream>>>(W_out, wl_out);
    layout_wa1_k<<<64, 256, 0, stream>>>(W_a1, wlay);

    // 2. node linear transforms (fp32 WMMA), 3125 tiles, 4 waves/block
    const int gemm_blocks = (3125 + 3) / 4;
    gemm128_f32_wmma<<<gemm_blocks, 128, 0, stream>>>(feat_src, wl_src, b_src,
                                                      hs, hsb, N_SRCC, 0);
    gemm128_f32_wmma<<<gemm_blocks, 128, 0, stream>>>(feat_dst, wl_dst, b_dst,
                                                      hd, hdb, N_DSTT, 0);

    // 3. edge attention scores (bf16 WMMA), 40000 tiles, 4 waves/block
    edge_scores_k<<<N_EDGE / 16 / 4, 128, 0, stream>>>(hsb, hdb, src_idx, dst_idx,
                                                       wlay, b_a1, W_a2, b_a2, sc);

    // 4. softmax over segments + weighted aggregation
    seg_max_k<<<(N_EDGE + 255) / 256, 256, 0, stream>>>(sc, dst_idx, mx);
    seg_exp_k<<<(N_EDGE + 255) / 256, 256, 0, stream>>>(sc, dst_idx, mx, ex, dn);
    edge_aggregate_k<<<(N_EDGE * 32) / 256, 256, 0, stream>>>(hs, hd, src_idx,
                                                              dst_idx, ex, dn, nf);

    // 5. output projection + leaky relu (fp32 WMMA) into d_out
    gemm128_f32_wmma<<<gemm_blocks, 128, 0, stream>>>(nf, wl_out, b_out,
                                                      (float*)d_out, nullptr,
                                                      N_DSTT, 1);
}
